// RNNModel_88639535055430
// MI455X (gfx1250) — compile-verified, matrix operand order checked by reading, not run on hardware
//
#include <hip/hip_runtime.h>

// ---------------------------------------------------------------------------
// Model dims (compile-time constants from the reference)
// ---------------------------------------------------------------------------
#define T_    128
#define B_    64
#define IN_D  512
#define NINP  1024
#define NHID  2048
#define NB    8
#define TOPK  5
#define MBLK  256          // M = NHID/NB
#define ATT_OUT 1024       // 4*MBLK
#define DK1   64
#define H2    4
#define DK2   32
#define OUT_D 10000

typedef __attribute__((ext_vector_type(16))) __bf16 v16bf;
typedef __attribute__((ext_vector_type(8)))  __bf16 v8bf;
typedef __attribute__((ext_vector_type(8)))  float  v8f;
typedef __attribute__((ext_vector_type(4)))  float  v4f;

__device__ __forceinline__ float sigmoidf(float x) { return 1.0f / (1.0f + __expf(-x)); }

// ---------------------------------------------------------------------------
// Generic batched GEMM:  C[z] (+)= A[z] (MxK, row stride lda) * W[z] (KxN) + bias[z]
// bf16 WMMA, f32 accumulation.  256 threads = 8 waves; each wave owns a 32x32
// C sub-tile (2x2 WMMA frags).  Tile = (WM*32) x (WN*32), WM*WN == 8.
// REQUIRES: K % 32 == 0 (true for every GEMM in this model).
// ---------------------------------------------------------------------------
#define BK 32

template<int WM, int WN>
__global__ void __launch_bounds__(256)
wmma_gemm(const float* __restrict__ A, long lda, long strideA,
          const float* __restrict__ W, long strideW,
          const float* __restrict__ bias, long strideBias,
          float* __restrict__ C, long ldc, long strideC,
          int M, int N, int K, int addC)
{
    constexpr int TBM = WM * 32;
    constexpr int TBN = WN * 32;
    constexpr int APT = TBM / 8;          // bf16 per thread for A tile (8 or 16)
    constexpr int TPR = 256 / TBM;        // threads per A row (2 or 4)
    constexpr int BPT = TBN / 8;          // bf16 per thread for B tile (8 or 16)

    __shared__ __align__(16) __bf16 sA[TBM * BK];   // [m][k]
    __shared__ __align__(16) __bf16 sB[TBN * BK];   // [n][k]  (transposed stage)

    const int tid   = threadIdx.x;
    const int lane  = tid & 31;
    const int wave  = tid >> 5;            // 0..7
    const int wm    = wave % WM;           // 32-row sub-block
    const int wn    = wave / WM;           // 32-col sub-block
    const int lmod  = lane & 15;
    const int lhalf = lane >> 4;

    const int rowBase = blockIdx.y * TBM;
    const int colBase = blockIdx.x * TBN;
    const long zb = blockIdx.z;

    const float* Ab    = A + zb * strideA;
    const float* Wb    = W + zb * strideW;
    const float* biasb = bias ? (bias + zb * strideBias) : nullptr;
    float*       Cb    = C + zb * strideC;

    v8f acc[2][2] = {};

    // A staging map: contiguous APT floats per thread
    const int aRow = tid / TPR;
    const int aK   = (tid % TPR) * APT;
    // B staging map: lane-coalesced along n, BPT k-values per thread
    const int bN   = tid % TBN;
    const int bKg  = (tid / TBN) * BPT;

    for (int k0 = 0; k0 < K; k0 += BK) {
        // ---- stage A tile (f32 -> bf16), row-major [m][k] ----
        {
            const int row = rowBase + aRow;
            __bf16* sa = &sA[aRow * BK + aK];
            if (row < M) {
                const float* ap = Ab + (size_t)row * lda + k0 + aK;
                if (k0 + BK < K) __builtin_prefetch(ap + BK, 0, 1);
#pragma unroll
                for (int u = 0; u < APT / 8; ++u) {
                    v4f f0 = ((const v4f*)ap)[2 * u + 0];
                    v4f f1 = ((const v4f*)ap)[2 * u + 1];
                    v8bf p;
#pragma unroll
                    for (int j = 0; j < 4; ++j) {
                        p[j]     = (__bf16)f0[j];
                        p[4 + j] = (__bf16)f1[j];
                    }
                    ((v8bf*)sa)[u] = p;
                }
            } else {
                v8bf z = {};
#pragma unroll
                for (int u = 0; u < APT / 8; ++u) ((v8bf*)sa)[u] = z;
            }
        }
        // ---- stage B tile transposed [n][k]; global reads coalesced in n ----
        {
            const int col = colBase + bN;
            __bf16* sb = &sB[bN * BK + bKg];
            if (col < N) {
                const float* wp = Wb + (size_t)(k0 + bKg) * N + col;
#pragma unroll
                for (int u = 0; u < BPT / 8; ++u) {
                    v8bf p;
#pragma unroll
                    for (int j = 0; j < 8; ++j)
                        p[j] = (__bf16)wp[(size_t)(u * 8 + j) * N];
                    ((v8bf*)sb)[u] = p;
                }
            } else {
                v8bf z = {};
#pragma unroll
                for (int u = 0; u < BPT / 8; ++u) ((v8bf*)sb)[u] = z;
            }
        }
        __syncthreads();

        // ---- gather per-lane fragments (16B LDS reads, wave32 layout) ----
        v16bf af[2], bfr[2];
#pragma unroll
        for (int i = 0; i < 2; ++i) {
            const __bf16* ap = &sA[(wm * 32 + i * 16 + lmod) * BK + lhalf * 8];
            v8bf alo = *(const v8bf*)ap;
            v8bf ahi = *(const v8bf*)(ap + 16);
#pragma unroll
            for (int j = 0; j < 8; ++j) { af[i][j] = alo[j]; af[i][j + 8] = ahi[j]; }

            const __bf16* bp = &sB[(wn * 32 + i * 16 + lmod) * BK + lhalf * 8];
            v8bf blo = *(const v8bf*)bp;
            v8bf bhi = *(const v8bf*)(bp + 16);
#pragma unroll
            for (int j = 0; j < 8; ++j) { bfr[i][j] = blo[j]; bfr[i][j + 8] = bhi[j]; }
        }

#pragma unroll
        for (int mi = 0; mi < 2; ++mi)
#pragma unroll
            for (int ni = 0; ni < 2; ++ni)
                acc[mi][ni] = __builtin_amdgcn_wmma_f32_16x16x32_bf16(
                    false, af[mi], false, bfr[ni], (short)0, acc[mi][ni], false, false);

        __syncthreads();
    }

    // ---- store (C layout: vgpr r -> row r, +8 for upper half-lanes) ----
#pragma unroll
    for (int mi = 0; mi < 2; ++mi) {
#pragma unroll
        for (int ni = 0; ni < 2; ++ni) {
            const int col = colBase + wn * 32 + ni * 16 + lmod;
            if (col >= N) continue;
            const float bv = biasb ? biasb[col] : 0.0f;
#pragma unroll
            for (int r = 0; r < 8; ++r) {
                const int row = rowBase + wm * 32 + mi * 16 + lhalf * 8 + r;
                if (row >= M) continue;
                const size_t off = (size_t)row * ldc + col;
                const float v = acc[mi][ni][r] + bv;
                if (addC) Cb[off] += v; else Cb[off] = v;
            }
        }
    }
}

// ---------------------------------------------------------------------------
// Input attention: 2-key softmax + value mix.  One block per (b, n).
// keys_in row0 is all-zero => k0 = bk1, v0 = bv1.
// ---------------------------------------------------------------------------
__global__ void attn1_kernel(const float* __restrict__ q1,     // (B*NB, 64)
                             const float* __restrict__ k1row,  // (B, 64)
                             const float* __restrict__ bk1,    // (64)
                             const float* __restrict__ v1row,  // (B, 1024)
                             const float* __restrict__ bv1,    // (1024)
                             float* __restrict__ att0,         // (B, NB)
                             float* __restrict__ inp_use)      // (B, NB, 1024)
{
    const int bn = blockIdx.x;
    const int b  = bn >> 3;
    __shared__ float red0[64], red1[64];
    __shared__ float sa0, sa1;
    const int tid = threadIdx.x;

    if (tid < 64) {
        const float qv = q1[bn * 64 + tid];
        red0[tid] = qv * bk1[tid];
        red1[tid] = qv * k1row[b * 64 + tid];
    }
    __syncthreads();
    if (tid == 0) {
        float l0 = 0.f, l1 = 0.f;
        for (int i = 0; i < 64; ++i) { l0 += red0[i]; l1 += red1[i]; }
        l0 *= 0.125f; l1 *= 0.125f;                 // 1/sqrt(DK1)
        const float mx = fmaxf(l0, l1);
        const float e0 = __expf(l0 - mx), e1 = __expf(l1 - mx);
        const float inv = 1.0f / (e0 + e1);
        sa0 = e0 * inv; sa1 = e1 * inv;
        att0[bn] = sa0;
    }
    __syncthreads();
    const float a0 = sa0, a1 = sa1;
    for (int g = tid; g < ATT_OUT; g += blockDim.x)
        inp_use[(size_t)bn * ATT_OUT + g] = a0 * bv1[g] + a1 * v1row[b * ATT_OUT + g];
}

// Drop the NB-TOPK blocks with the largest null-key attention.
__global__ void topk_mask_kernel(const float* __restrict__ att0, float* __restrict__ mask)
{
    const int b = blockIdx.x;
    if (threadIdx.x != 0) return;
    float a[NB], m[NB];
    for (int i = 0; i < NB; ++i) { a[i] = att0[b * NB + i]; m[i] = 1.0f; }
    for (int s = 0; s < NB - TOPK; ++s) {
        int arg = 0; float best = a[0];
        for (int i = 1; i < NB; ++i) if (a[i] > best) { best = a[i]; arg = i; }
        m[arg] = 0.0f; a[arg] = -1e30f;
    }
    for (int i = 0; i < NB; ++i) mask[b * NB + i] = m[i];
}

// LSTM gates: pre (B,NB,4*MBLK) split i|f|g|o
__global__ void lstm_kernel(const float* __restrict__ pre, const float* __restrict__ cx,
                            float* __restrict__ c_new, float* __restrict__ h_new)
{
    const int idx = blockIdx.x * blockDim.x + threadIdx.x;
    if (idx >= B_ * NB * MBLK) return;
    const int m  = idx & (MBLK - 1);
    const int bn = idx >> 8;
    const float* p = pre + (size_t)bn * (4 * MBLK);
    const float ig = p[m], fg = p[MBLK + m], gg = p[2 * MBLK + m], og = p[3 * MBLK + m];
    const float c = sigmoidf(fg) * cx[idx] + sigmoidf(ig) * tanhf(gg);
    c_new[idx] = c;
    h_new[idx] = sigmoidf(og) * tanhf(c);
}

// Block-to-block attention: 8x8 softmax per (b, head)
__global__ void attn2_kernel(const float* __restrict__ q2, const float* __restrict__ k2,
                             const float* __restrict__ v2, float* __restrict__ o2)
{
    const int bh = blockIdx.x;
    const int b = bh >> 2, h = bh & 3;
    __shared__ float lg[NB][NB];
    __shared__ float at[NB][NB];
    const int tid = threadIdx.x;          // 64 threads
    const int n = tid >> 3, mm = tid & 7;

    const float* qp = q2 + ((size_t)(b * NB + n)) * (H2 * DK2) + h * DK2;
    const float* kp = k2 + ((size_t)(b * NB + mm)) * (H2 * DK2) + h * DK2;
    float s = 0.f;
    for (int d = 0; d < DK2; ++d) s += qp[d] * kp[d];
    lg[n][mm] = s * 0.17677669529663687f; // 1/sqrt(DK2)
    __syncthreads();

    if (tid < NB) {
        float mx = lg[tid][0];
        for (int j = 1; j < NB; ++j) mx = fmaxf(mx, lg[tid][j]);
        float e[NB], sum = 0.f;
        for (int j = 0; j < NB; ++j) { e[j] = __expf(lg[tid][j] - mx); sum += e[j]; }
        const float inv = 1.0f / sum;
        for (int j = 0; j < NB; ++j) at[tid][j] = e[j] * inv;
    }
    __syncthreads();

    for (int r = tid; r < NB * DK2; r += 64) {
        const int nn = r >> 5, d = r & 31;
        float acc = 0.f;
        for (int j = 0; j < NB; ++j)
            acc += at[nn][j] * v2[((size_t)(b * NB + j)) * (H2 * DK2) + h * DK2 + d];
        o2[((size_t)(b * NB + nn)) * (H2 * DK2) + h * DK2 + d] = acc;
    }
}

// comm gate + masked hx/cx blend + record outs[t]
__global__ void update_kernel(const float* __restrict__ f_out, const float* __restrict__ g_out,
                              const float* __restrict__ h_new, const float* __restrict__ c_new,
                              const float* __restrict__ mask,
                              float* __restrict__ hx, float* __restrict__ cx,
                              float* __restrict__ out_t)
{
    const int idx = blockIdx.x * blockDim.x + threadIdx.x;
    if (idx >= B_ * NHID) return;
    const int bn = idx >> 8;                       // (b*NB + n)
    const float comm = sigmoidf(g_out[idx]) * tanhf(f_out[idx]);
    const float hf = h_new[idx] + comm;
    const float mf = mask[bn];
    const float ho = mf * hf + (1.0f - mf) * hx[idx];
    const float co = mf * c_new[idx] + (1.0f - mf) * cx[idx];
    hx[idx] = ho; cx[idx] = co; out_t[idx] = ho;
}

// ---------------------------------------------------------------------------
static inline void gemm(hipStream_t s,
                        const float* A, long lda, long strideA,
                        const float* W, long strideW,
                        const float* bias, long strideBias,
                        float* C, long ldc, long strideC,
                        int M, int N, int K, int batch, int addC)
{
    if (M >= 128) {
        dim3 g((N + 63) / 64, (M + 127) / 128, batch);
        wmma_gemm<4, 2><<<g, 256, 0, s>>>(A, lda, strideA, W, strideW, bias, strideBias,
                                          C, ldc, strideC, M, N, K, addC);
    } else {
        dim3 g((N + 127) / 128, (M + 63) / 64, batch);
        wmma_gemm<2, 4><<<g, 256, 0, s>>>(A, lda, strideA, W, strideW, bias, strideBias,
                                          C, ldc, strideC, M, N, K, addC);
    }
}

extern "C" void kernel_launch(void* const* d_in, const int* in_sizes, int n_in,
                              void* d_out, int out_size, void* d_ws, size_t ws_size,
                              hipStream_t stream)
{
    const float* x    = (const float*)d_in[0];
    const float* hx0  = (const float*)d_in[1];
    const float* cx0  = (const float*)d_in[2];
    const float* encW = (const float*)d_in[3];
    const float* encb = (const float*)d_in[4];
    const float* Wq1  = (const float*)d_in[5];
    const float* bq1  = (const float*)d_in[6];
    const float* Wk1  = (const float*)d_in[7];
    const float* bk1  = (const float*)d_in[8];
    const float* Wv1  = (const float*)d_in[9];
    const float* bv1  = (const float*)d_in[10];
    const float* Wi   = (const float*)d_in[11];
    const float* Wh   = (const float*)d_in[12];
    const float* bi   = (const float*)d_in[13];
    const float* bh   = (const float*)d_in[14];
    const float* Wq2  = (const float*)d_in[15];
    const float* bq2  = (const float*)d_in[16];
    const float* Wk2  = (const float*)d_in[17];
    const float* bk2  = (const float*)d_in[18];
    const float* Wv2  = (const float*)d_in[19];
    const float* bv2  = (const float*)d_in[20];
    const float* Wf   = (const float*)d_in[21];
    const float* bfp  = (const float*)d_in[22];
    const float* Wg   = (const float*)d_in[23];
    const float* bgp  = (const float*)d_in[24];
    const float* decW = (const float*)d_in[25];
    const float* decb = (const float*)d_in[26];
    float* out = (float*)d_out;

    // ---- workspace carve-up (floats) ----
    float* ws = (float*)d_ws;
    size_t o = 0;
    float* emb   = ws + o; o += (size_t)T_ * B_ * NINP;      // 8.4M
    float* outs  = ws + o; o += (size_t)T_ * B_ * NHID;      // 16.8M
    float* hx    = ws + o; o += (size_t)B_ * NHID;
    float* cx    = ws + o; o += (size_t)B_ * NHID;
    float* q1b   = ws + o; o += (size_t)B_ * NB * DK1;
    float* k1b   = ws + o; o += (size_t)B_ * DK1;
    float* v1b   = ws + o; o += (size_t)B_ * ATT_OUT;
    float* att0  = ws + o; o += (size_t)B_ * NB;
    float* maskb = ws + o; o += (size_t)B_ * NB;
    float* inp   = ws + o; o += (size_t)B_ * NB * ATT_OUT;
    float* pre   = ws + o; o += (size_t)B_ * NB * 4 * MBLK;
    float* cnew  = ws + o; o += (size_t)B_ * NB * MBLK;
    float* hnew  = ws + o; o += (size_t)B_ * NB * MBLK;
    float* q2b   = ws + o; o += (size_t)B_ * NB * H2 * DK2;
    float* k2b   = ws + o; o += (size_t)B_ * NB * H2 * DK2;
    float* v2b   = ws + o; o += (size_t)B_ * NB * H2 * DK2;
    float* o2b   = ws + o; o += (size_t)B_ * NB * H2 * DK2;
    float* fob   = ws + o; o += (size_t)B_ * NB * MBLK;
    float* gob   = ws + o; o += (size_t)B_ * NB * MBLK;

    const size_t stateBytes = (size_t)B_ * NHID * sizeof(float);
    hipMemcpyAsync(hx, hx0, stateBytes, hipMemcpyDeviceToDevice, stream);
    hipMemcpyAsync(cx, cx0, stateBytes, hipMemcpyDeviceToDevice, stream);

    // encoder: emb = x @ enc_W + enc_b    (8192 x 512) x (512 x 1024)
    gemm(stream, x, IN_D, 0, encW, 0, encb, 0, emb, NINP, 0,
         T_ * B_, NINP, IN_D, 1, 0);

    for (int t = 0; t < T_; ++t) {
        const float* et = emb + (size_t)t * B_ * NINP;
        float* ot = outs + (size_t)t * B_ * NHID;

        // q = hblk @ Wq1 + bq1 : view hx as (B*NB, MBLK)
        gemm(stream, hx, MBLK, 0, Wq1, 0, bq1, 0, q1b, DK1, 0,
             B_ * NB, DK1, MBLK, 1, 0);
        // k row-1 / v row-1 (row-0 of keys_in is zero => bias only)
        gemm(stream, et, NINP, 0, Wk1, 0, bk1, 0, k1b, DK1, 0,
             B_, DK1, NINP, 1, 0);
        gemm(stream, et, NINP, 0, Wv1, 0, bv1, 0, v1b, ATT_OUT, 0,
             B_, ATT_OUT, NINP, 1, 0);

        attn1_kernel<<<B_ * NB, 256, 0, stream>>>(q1b, k1b, bk1, v1b, bv1, att0, inp);
        topk_mask_kernel<<<B_, 32, 0, stream>>>(att0, maskb);

        // pre = inp_use @ Wi + bi  (batched over NB blocks), 64x128 tiles
        gemm(stream, inp, (long)NB * ATT_OUT, ATT_OUT,
             Wi, (long)ATT_OUT * 4 * MBLK, bi, 4 * MBLK,
             pre, (long)NB * 4 * MBLK, 4 * MBLK,
             B_, 4 * MBLK, ATT_OUT, NB, 0);
        // pre += hblk @ Wh + bh
        gemm(stream, hx, NHID, MBLK,
             Wh, (long)MBLK * 4 * MBLK, bh, 4 * MBLK,
             pre, (long)NB * 4 * MBLK, 4 * MBLK,
             B_, 4 * MBLK, MBLK, NB, 1);

        lstm_kernel<<<(B_ * NB * MBLK + 255) / 256, 256, 0, stream>>>(pre, cx, cnew, hnew);

        // q2/k2/v2 = h_new @ W*2 + b*2 : (512 x 256) x (256 x 128)
        gemm(stream, hnew, MBLK, 0, Wq2, 0, bq2, 0, q2b, H2 * DK2, 0,
             B_ * NB, H2 * DK2, MBLK, 1, 0);
        gemm(stream, hnew, MBLK, 0, Wk2, 0, bk2, 0, k2b, H2 * DK2, 0,
             B_ * NB, H2 * DK2, MBLK, 1, 0);
        gemm(stream, hnew, MBLK, 0, Wv2, 0, bv2, 0, v2b, H2 * DK2, 0,
             B_ * NB, H2 * DK2, MBLK, 1, 0);

        attn2_kernel<<<B_ * H2, 64, 0, stream>>>(q2b, k2b, v2b, o2b);

        // comm projections: (512 x 128) x (128 x 256)
        gemm(stream, o2b, H2 * DK2, 0, Wf, 0, bfp, 0, fob, MBLK, 0,
             B_ * NB, MBLK, H2 * DK2, 1, 0);
        gemm(stream, o2b, H2 * DK2, 0, Wg, 0, bgp, 0, gob, MBLK, 0,
             B_ * NB, MBLK, H2 * DK2, 1, 0);

        update_kernel<<<(B_ * NHID + 255) / 256, 256, 0, stream>>>(
            fob, gob, hnew, cnew, maskb, hx, cx, ot);
    }

    // decoder: dec = outs @ dec_W + dec_b  (8192 x 2048) x (2048 x 10000)
    gemm(stream, outs, NHID, 0, decW, 0, decb, 0, out, OUT_D, 0,
         T_ * B_, OUT_D, NHID, 1, 0);

    // append hx_f, cx_f after dec in the flat output
    float* tail = out + (size_t)T_ * B_ * OUT_D;
    hipMemcpyAsync(tail, hx, stateBytes, hipMemcpyDeviceToDevice, stream);
    hipMemcpyAsync(tail + (size_t)B_ * NHID, cx, stateBytes, hipMemcpyDeviceToDevice, stream);
}
